// Scaled_Dot_Product_Attention_69114613727753
// MI455X (gfx1250) — matrix operand order repaired and checked by reading, not compiled, not run on hardware
//
#include <hip/hip_runtime.h>

typedef __attribute__((ext_vector_type(16))) _Float16 v16h;
typedef __attribute__((ext_vector_type(8)))  _Float16 v8h;
typedef __attribute__((ext_vector_type(2)))  _Float16 v2h;
typedef __attribute__((ext_vector_type(8)))  float    v8f;
typedef __attribute__((ext_vector_type(4)))  float    v4f;

#define B_    16
#define S_    2048
#define D_    128
#define BN    32            // KV tile rows
#define WAVES 4
#define BM    (WAVES * 16)  // Q rows per workgroup

// ---- DPP16 butterfly max over a 16-lane group (pure VALU, no LDS).
// ^1(quad_perm), ^2(quad_perm), row_half_mirror(quad^1), row_mirror(quad^3)
// covers all 16 lanes; DPP16 rows are 16 lanes so lane groups 0-15 / 16-31
// reduce independently (matching the C-fragment row grouping).
template <int CTRL>
__device__ __forceinline__ float dpp_xchg(float x) {
    return __int_as_float(__builtin_amdgcn_update_dpp(
        0, __float_as_int(x), CTRL, 0xF, 0xF, true));
}
__device__ __forceinline__ float rowmax16(float x) {
    x = fmaxf(x, dpp_xchg<0xB1>(x));   // quad_perm [1,0,3,2]  : ^1
    x = fmaxf(x, dpp_xchg<0x4E>(x));   // quad_perm [2,3,0,1]  : ^2
    x = fmaxf(x, dpp_xchg<0x141>(x));  // row_half_mirror      : quad^1
    x = fmaxf(x, dpp_xchg<0x140>(x));  // row_mirror           : quad^3
    return x;
}

// Flash-attention-2 for CDNA5 (gfx1250), wave32, v_wmma_f32_16x16x32_f16.
// Each wave owns a 16-row Q tile; KV streamed in 32-row tiles via LDS (f16),
// software-pipelined (next tile's global loads + f16 converts issue before /
// during current compute; staging is f16 to keep allocation <= 256 VGPRs).
// The softmax denominator runs on the matrix pipe: an extra WMMA against an
// all-ones B fragment yields rowsum(P) directly in C-fragment layout.
__launch_bounds__(128)
__global__ void fa2_cdna5_kernel(const float* __restrict__ Q,
                                 const float* __restrict__ K,
                                 const float* __restrict__ V,
                                 float* __restrict__ O) {
    __shared__ __align__(64) _Float16 Kh[BN][D_];        // K tile, row-major f16   (8 KB)
    __shared__ __align__(64) _Float16 Vt[D_][BN];        // V tile, transposed f16  (8 KB)
    __shared__ __align__(64) _Float16 Ps[WAVES][16][BN]; // per-wave P relayout     (4 KB)

    const int tid  = threadIdx.x;
    const int wave = tid >> 5;
    const int lane = tid & 31;
    const int n    = lane & 15;   // fragment column (B/C) or row (A)
    const int g    = lane >> 4;   // lane group

    const int ntile_m = S_ / BM;
    const int b     = blockIdx.x / ntile_m;
    const int mtile = blockIdx.x % ntile_m;
    const int qbase = mtile * BM + wave * 16;

    // fold 1/sqrt(D) and log2(e) into Q so softmax uses exp2
    const float qscale = 0.08838834764831845f * 1.4426950408889634f;

    // ---- Q tile as 4 A-fragments (16x32 f16), k order shared with K/P/V fragments:
    // half h -> K = h + 16*g (contiguous)
    v16h qa[4];
    {
        const float* qrow = Q + (size_t)(b * S_ + qbase + n) * D_;
        #pragma unroll
        for (int c = 0; c < 4; ++c) {
            const float* p0 = qrow + c * 32 + 16 * g;
            const float* p1 = qrow + c * 32 + 8 + 16 * g;
            v4f a0 = *(const v4f*)(p0);
            v4f a1 = *(const v4f*)(p0 + 4);
            v4f b0 = *(const v4f*)(p1);
            v4f b1 = *(const v4f*)(p1 + 4);
            #pragma unroll
            for (int i = 0; i < 4; ++i) {
                qa[c][i]      = (_Float16)(a0[i] * qscale);
                qa[c][i + 4]  = (_Float16)(a1[i] * qscale);
                qa[c][i + 8]  = (_Float16)(b0[i] * qscale);
                qa[c][i + 12] = (_Float16)(b1[i] * qscale);
            }
        }
    }

    // all-ones B fragment: one extra WMMA per tile produces rowsum(P)
    v16h ones;
    #pragma unroll
    for (int i = 0; i < 16; ++i) ones[i] = (_Float16)1.0f;

    v8f o_acc[8] = {};   // 16x128 fp32 accumulator (8 C-fragments)
    v8f l_acc = {};      // softmax denominator, C-fragment layout (row = r+8g)
    float m_i[8];        // per-row running max
    #pragma unroll
    for (int r = 0; r < 8; ++r) m_i[r] = -3.0e38f;

    // cooperative loader: each thread owns a KV row-pair x 16 columns
    const int rp = tid >> 3;          // 0..15 row pair
    const int r0 = rp * 2;
    const int cb = (tid & 7) * 16;    // column base

    // f16 staging regs (32 VGPRs): K as two rows x 16 cols, V pre-packed in
    // the transposed row-pair format (maps onto v_cvt_pk_f16_f32 at convert)
    v8h kh[4];    // kh[rr*2+q] = Kh[r0+rr][cb+8q .. +7]
    v2h vp[16];   // vp[i]      = { V[r0][cb+i], V[r0+1][cb+i] }
    auto load_tile = [&](int jb) {
        const float* kr = K + (size_t)(b * S_ + jb + r0) * D_ + cb;
        const float* vr = V + (size_t)(b * S_ + jb + r0) * D_ + cb;
        #pragma unroll
        for (int q = 0; q < 2; ++q) {
            v4f x0 = *(const v4f*)(kr + q * 8);
            v4f x1 = *(const v4f*)(kr + q * 8 + 4);
            v4f y0 = *(const v4f*)(kr + D_ + q * 8);
            v4f y1 = *(const v4f*)(kr + D_ + q * 8 + 4);
            #pragma unroll
            for (int i = 0; i < 4; ++i) {
                kh[q][i]         = (_Float16)x0[i];
                kh[q][i + 4]     = (_Float16)x1[i];
                kh[2 + q][i]     = (_Float16)y0[i];
                kh[2 + q][i + 4] = (_Float16)y1[i];
            }
        }
        #pragma unroll
        for (int q = 0; q < 4; ++q) {
            v4f a = *(const v4f*)(vr + q * 4);
            v4f c = *(const v4f*)(vr + D_ + q * 4);
            #pragma unroll
            for (int i = 0; i < 4; ++i) {
                vp[q * 4 + i][0] = (_Float16)a[i];
                vp[q * 4 + i][1] = (_Float16)c[i];
            }
        }
    };
    auto store_tile = [&]() {     // pure DS stores: 4x b128 + 16x b32
        *(v8h*)&Kh[r0][cb]         = kh[0];
        *(v8h*)&Kh[r0][cb + 8]     = kh[1];
        *(v8h*)&Kh[r0 + 1][cb]     = kh[2];
        *(v8h*)&Kh[r0 + 1][cb + 8] = kh[3];
        #pragma unroll
        for (int i = 0; i < 16; ++i)
            *(v2h*)&Vt[cb + i][r0] = vp[i];
    };

    load_tile(0);   // prologue

    for (int jbase = 0; jbase < S_; jbase += BN) {
        __syncthreads();        // previous tile's LDS readers done
        store_tile();           // staged f16 tile -> LDS
        __syncthreads();
        if (jbase + BN < S_)    // issue next tile's loads before compute
            load_tile(jbase + BN);

        // ---- S = (Q*scale) K^T : two 16x16 tiles over 32 kv rows, 4 K-chunks
        // B-layout: lane col = n, half h -> K = h + 16*g (contiguous in Kh row)
        v8f s0 = {}, s1 = {};
        #pragma unroll
        for (int c = 0; c < 4; ++c) {
            v16h kb0 = *(const v16h*)&Kh[n][c * 32 + 16 * g];
            v16h kb1 = *(const v16h*)&Kh[n + 16][c * 32 + 16 * g];
            s0 = __builtin_amdgcn_wmma_f32_16x16x32_f16(false, qa[c], false, kb0,
                                                        (short)0, s0, false, false);
            s1 = __builtin_amdgcn_wmma_f32_16x16x32_f16(false, qa[c], false, kb1,
                                                        (short)0, s1, false, false);
        }

        // ---- online softmax (base-2); C-layout row r+8g lives in a 16-lane group
        #pragma unroll
        for (int r = 0; r < 8; ++r) {
            float mx    = rowmax16(fmaxf(s0[r], s1[r]));
            float mnew  = fmaxf(m_i[r], mx);
            float alpha = __builtin_amdgcn_exp2f(m_i[r] - mnew);
            m_i[r] = mnew;
            float p0 = __builtin_amdgcn_exp2f(s0[r] - mnew);
            float p1 = __builtin_amdgcn_exp2f(s1[r] - mnew);
            l_acc[r] *= alpha;                 // denom rescale (rowsum added by WMMA)
            #pragma unroll
            for (int ch = 0; ch < 8; ++ch) o_acc[ch][r] *= alpha;
            // write P tile (f16) into per-wave scratch for C->A relayout
            Ps[wave][r + 8 * g][n]      = (_Float16)p0;
            Ps[wave][r + 8 * g][n + 16] = (_Float16)p1;
        }
        // same-wave LDS RAW: DS ops are in-order per wave; block compiler reordering
        asm volatile("s_wait_dscnt 0" ::: "memory");

        // ---- reload P as a 16x32 A-fragment
        v16h pa;
        {
            v8h lo = *(const v8h*)&Ps[wave][n][16 * g];      // K = 16g..16g+7
            v8h hi = *(const v8h*)&Ps[wave][n][8 + 16 * g];  // K = 16g+8..+7
            #pragma unroll
            for (int i = 0; i < 8; ++i) { pa[i] = lo[i]; pa[i + 8] = hi[i]; }
        }

        // ---- denominator on the matrix pipe: l += P * ones (rowsum in C layout)
        l_acc = __builtin_amdgcn_wmma_f32_16x16x32_f16(false, pa, false, ones,
                                                       (short)0, l_acc, false, false);

        // ---- O += P V : 8 column chunks of 16, K-dim = 32 (one WMMA each)
        #pragma unroll
        for (int ch = 0; ch < 8; ++ch) {
            v16h vb = *(const v16h*)&Vt[ch * 16 + n][16 * g];
            o_acc[ch] = __builtin_amdgcn_wmma_f32_16x16x32_f16(false, pa, false, vb,
                                                               (short)0, o_acc[ch], false, false);
        }
    }

    // ---- normalize by softmax denominator and store fp32 output
    #pragma unroll
    for (int r = 0; r < 8; ++r) {
        float inv = 1.0f / l_acc[r];
        const int row = qbase + r + 8 * g;
        float* orow = O + (size_t)(b * S_ + row) * D_;
        #pragma unroll
        for (int ch = 0; ch < 8; ++ch)
            orow[ch * 16 + n] = o_acc[ch][r] * inv;
    }
}

extern "C" void kernel_launch(void* const* d_in, const int* in_sizes, int n_in,
                              void* d_out, int out_size, void* d_ws, size_t ws_size,
                              hipStream_t stream) {
    (void)in_sizes; (void)n_in; (void)d_ws; (void)ws_size; (void)out_size;
    const float* Q = (const float*)d_in[0];
    const float* K = (const float*)d_in[1];
    const float* V = (const float*)d_in[2];
    float* O = (float*)d_out;

    dim3 grid(B_ * (S_ / BM));   // 16 * 32 = 512 workgroups
    dim3 block(WAVES * 32);      // 128 threads = 4 waves (wave32)
    fa2_cdna5_kernel<<<grid, block, 0, stream>>>(Q, K, V, O);
}
